// MyModel_73478300500355
// MI455X (gfx1250) — matrix operand order repaired and checked by reading, not compiled, not run on hardware
//
#include <hip/hip_runtime.h>
#include <hip/hip_bf16.h>
#include <stdint.h>

// Problem constants
#define BSZ 256
#define TLEN 512
#define EDIM 256
#define UDIM 1024
#define U3 3072
#define VOC 128

// Tiling: block = 4 waves (wave32), wave tile 16x64, block tile 32x128, BK=32
#define BDIM 128
#define BM 32
#define BN 128
#define BK 32
#define AS 40   // LDS A row stride in ushorts: 80B = 5*16B -> b128-aligned rows
#define BS 40   // LDS B^T col stride in ushorts
#define NWG_SCAN ((UDIM / BN) * (BSZ / BM))   // 64 persistent workgroups

typedef __attribute__((ext_vector_type(16))) __bf16 v16bf;
typedef __attribute__((ext_vector_type(8)))  float  v8f;

union Frag16 { uint32_t u[8]; v16bf v; };
union FragC  { float    f[8]; v8f   v; };

__device__ __forceinline__ uint16_t f2bf(float f) {
  uint32_t x = __float_as_uint(f);
  uint32_t r = x + 0x7FFFu + ((x >> 16) & 1u);  // round-to-nearest-even
  return (uint16_t)(r >> 16);
}

// Low 32 bits of a generic LDS pointer = wave-relative LDS byte offset.
__device__ __forceinline__ uint32_t lds_off(const void* p) {
  return (uint32_t)(uintptr_t)p;
}

// gfx1250 async DMA: 16B global -> LDS, tracked by ASYNCcnt (ISA 10 / 15.18.3).
__device__ __forceinline__ void async_b128(uint32_t lds_dst, const void* sbase,
                                           uint32_t voff) {
  asm volatile("global_load_async_to_lds_b128 %0, %1, %2"
               :: "v"(lds_dst), "v"(voff), "s"(sbase)
               : "memory");
}
__device__ __forceinline__ void wait_async0() {
  asm volatile("s_wait_asynccnt 0x0" ::: "memory");
}

// A fragment (16x32 bf16) from row-major LDS tile (contiguous k order: k = 16*kh + s).
__device__ __forceinline__ void load_afrag(const uint16_t* tileRowBase, Frag16& a) {
  const int lane = threadIdx.x & 31;
  const int m = lane & 15, kh = lane >> 4;
  const uint16_t* base = tileRowBase + m * AS;
#pragma unroll
  for (int v = 0; v < 8; ++v) {
    const int kb = 2 * v + 16 * kh;
    a.u[v] = *(const uint32_t*)(base + kb);
  }
}

// B fragment (32x16 bf16) from K-contiguous LDS tile.
__device__ __forceinline__ void load_bfrag(const uint16_t* tileColBase, Frag16& b) {
  const int lane = threadIdx.x & 31;
  const int n = lane & 15, kh = lane >> 4;
  const uint16_t* base = tileColBase + n * BS + 16 * kh;
#pragma unroll
  for (int v = 0; v < 8; ++v) b.u[v] = *(const uint32_t*)(base + 2 * v);
}

__device__ __forceinline__ void wmma_bf16(FragC& c, const Frag16& a, const Frag16& b) {
  c.v = __builtin_amdgcn_wmma_f32_16x16x32_bf16(false, a.v, false, b.v,
                                                (short)0, c.v, false, false);
}

// Device-scope sense-reversing barrier for the persistent scan kernel.
__device__ __forceinline__ void grid_barrier(uint32_t* cnt, uint32_t* gen) {
  __threadfence();                 // release: my stores reach device scope (L2)
  __syncthreads();                 // whole WG fenced before arrival
  if (threadIdx.x == 0) {
    const uint32_t g =
        __hip_atomic_load(gen, __ATOMIC_RELAXED, __HIP_MEMORY_SCOPE_AGENT);
    const uint32_t a =
        __hip_atomic_fetch_add(cnt, 1u, __ATOMIC_ACQ_REL, __HIP_MEMORY_SCOPE_AGENT);
    if (a == (uint32_t)NWG_SCAN - 1u) {
      __hip_atomic_store(cnt, 0u, __ATOMIC_RELAXED, __HIP_MEMORY_SCOPE_AGENT);
      __hip_atomic_store(gen, g + 1u, __ATOMIC_RELEASE, __HIP_MEMORY_SCOPE_AGENT);
    } else {
      while (__hip_atomic_load(gen, __ATOMIC_ACQUIRE,
                               __HIP_MEMORY_SCOPE_AGENT) == g) {
        __builtin_amdgcn_s_sleep(2);
      }
    }
  }
  __syncthreads();
  __threadfence();                 // acquire: invalidate stale WGP$ lines
}

// ---------------------------------------------------------------------------
__global__ void cvt_bf16_kernel(const float* __restrict__ in,
                                uint16_t* __restrict__ out, int n) {
  int i = blockIdx.x * blockDim.x + threadIdx.x;
  if (i < n) out[i] = f2bf(in[i]);
}

// fp32 [K][N] -> bf16 transposed [N][K] (one-time; makes staging contiguous)
__global__ void cvt_bf16_T_kernel(const float* __restrict__ in,
                                  uint16_t* __restrict__ out, int K, int N) {
  int i = blockIdx.x * blockDim.x + threadIdx.x;
  if (i < K * N) {
    int k = i / N, n = i % N;
    out[(size_t)n * K + k] = f2bf(in[i]);
  }
}

// ids [B][T] -> idsT [T][B] (contiguous per-step gather)
__global__ void transpose_ids_kernel(const int* __restrict__ ids,
                                     int* __restrict__ idsT) {
  int i = blockIdx.x * blockDim.x + threadIdx.x;
  if (i < BSZ * TLEN) {
    int b = i / TLEN, t = i % TLEN;
    idsT[t * BSZ + b] = ids[i];
  }
}

__global__ void zero_state_kernel(float* __restrict__ h32,
                                  uint16_t* __restrict__ hbf,
                                  uint32_t* __restrict__ bar, int n) {
  int i = blockIdx.x * blockDim.x + threadIdx.x;
  if (i < n) { h32[i] = 0.0f; hbf[i] = 0; }
  if (i == 0) { bar[0] = 0u; bar[1] = 0u; }
}

// ---------------------------------------------------------------------------
// Generic GEMM: C[M,N] = A[M,K](bf16 row-major) * B^T[N,K](bf16) + bias[N]
// grid = (N/BN, M/BM). Async-DMA staged, double-buffered LDS.
__global__ __launch_bounds__(BDIM) void gemm_bf16_kernel(
    const uint16_t* __restrict__ A, int lda,
    const uint16_t* __restrict__ BT, int ldbt,   // ldbt == K
    const float* __restrict__ bias,
    float* __restrict__ C, int ldc, int K) {
  __shared__ uint16_t ldsA[2 * BM * AS];
  __shared__ uint16_t ldsB[2 * BN * BS];

  const int tid = threadIdx.x;
  const int wave = tid >> 5;
  const int wm = wave >> 1, wn = wave & 1;
  const int m0 = blockIdx.y * BM;
  const int n0 = blockIdx.x * BN;
  const uint32_t ldsA_o = lds_off(ldsA);
  const uint32_t ldsB_o = lds_off(ldsB);
  const int ar = tid >> 2, ac = tid & 3;

  auto stage = [&](int k0, int buf) {
    async_b128(ldsA_o + (uint32_t)(((buf * BM + ar) * AS + ac * 8) * 2),
               A, (uint32_t)(((m0 + ar) * (size_t)lda + k0 + ac * 8) * 2));
#pragma unroll
    for (int i = 0; i < 4; ++i) {
      const int cid = tid + i * BDIM;
      const int n = cid >> 2, c = cid & 3;
      async_b128(ldsB_o + (uint32_t)(((buf * BN + n) * BS + c * 8) * 2),
                 BT, (uint32_t)(((n0 + n) * (size_t)ldbt + k0 + c * 8) * 2));
    }
  };

  FragC acc[4];
#pragma unroll
  for (int f = 0; f < 4; ++f)
#pragma unroll
    for (int v = 0; v < 8; ++v) acc[f].f[v] = 0.0f;

  const int nk = K / BK;
  stage(0, 0);
  wait_async0();
  __syncthreads();

  for (int kt = 0; kt < nk; ++kt) {
    const int cur = kt & 1;
    if (kt + 1 < nk) stage((kt + 1) * BK, cur ^ 1);

    Frag16 a;
    load_afrag(ldsA + (cur * BM + wm * 16) * AS, a);
#pragma unroll
    for (int f = 0; f < 4; ++f) {
      Frag16 b;
      load_bfrag(ldsB + (cur * BN + wn * 64 + f * 16) * BS, b);
      wmma_bf16(acc[f], a, b);
    }
    wait_async0();
    __syncthreads();
  }

  const int lane = tid & 31;
  const int n_l = lane & 15, kh = lane >> 4;
#pragma unroll
  for (int f = 0; f < 4; ++f) {
    const int col = n0 + wn * 64 + f * 16 + n_l;
    const float bv = bias[col];
#pragma unroll
    for (int v = 0; v < 8; ++v) {
      const int row = m0 + wm * 16 + v + 8 * kh;
      C[(size_t)row * ldc + col] = acc[f].f[v] + bv;
    }
  }
}

// ---------------------------------------------------------------------------
// Persistent GRU scan: all 512 timesteps in ONE launch (removes 512 launch
// overheads). 64 co-resident WGs, device-scope barrier per step, h double-
// buffered by parity. Per step/WG: async-DMA staged tiles, 12 WMMA per wave/k.
__global__ __launch_bounds__(BDIM) void gru_scan_kernel(
    uint16_t* __restrict__ hbf0, uint16_t* __restrict__ hbf1,  // [B,U] bf16
    float* __restrict__ h0, float* __restrict__ h1,            // [B,U] fp32
    const uint16_t* __restrict__ UhT,      // [3U,U] bf16 (pre-transposed)
    const float*    __restrict__ b_rec,    // [3U]
    const float*    __restrict__ P,        // [V,3U] = emb@Wx + b_in
    const int*      __restrict__ idsT,     // [T,B]
    uint16_t* __restrict__ hs,             // [B,T,U] bf16
    uint32_t* __restrict__ bar) {          // {count, generation}
  __shared__ uint16_t ldsA[2 * BM * AS];
  __shared__ uint16_t ldsB[2 * 3 * BN * BS];

  const int tid = threadIdx.x;
  const int wave = tid >> 5;
  const int wm = wave >> 1, wn = wave & 1;
  const int m0 = blockIdx.y * BM;  // batch rows
  const int n0 = blockIdx.x * BN;  // u columns
  const uint32_t ldsA_o = lds_off(ldsA);
  const uint32_t ldsB_o = lds_off(ldsB);
  const int ar = tid >> 2, ac = tid & 3;
  const int lane = tid & 31;
  const int n_l = lane & 15, kh = lane >> 4;

  for (int t = 0; t < TLEN; ++t) {
    const uint16_t* hbf_in  = (t & 1) ? hbf1 : hbf0;
    const float*    h_in    = (t & 1) ? h1 : h0;
    float*          h_out   = (t & 1) ? h0 : h1;
    uint16_t*       hbf_out = (t & 1) ? hbf0 : hbf1;

    auto stage = [&](int k0, int buf) {
      // A: h rows, BM x BK (one b128 per thread)
      async_b128(ldsA_o + (uint32_t)(((buf * BM + ar) * AS + ac * 8) * 2),
                 hbf_in, (uint32_t)((((m0 + ar) * UDIM) + k0 + ac * 8) * 2));
      // B: 3 gates x BN rows x BK from pre-transposed UhT (12 b128 per thread)
#pragma unroll
      for (int i = 0; i < 12; ++i) {
        const int cid = tid + i * BDIM;
        const int g = cid >> 9;
        const int rem = cid & 511;
        const int n = rem >> 2, c = rem & 3;
        async_b128(
            ldsB_o + (uint32_t)((((buf * 3 + g) * BN + n) * BS + c * 8) * 2),
            UhT, (uint32_t)(((g * UDIM + n0 + n) * UDIM + k0 + c * 8) * 2));
      }
      if (k0 + BK < UDIM)  // gfx1250 global_prefetch_b8, next tile into L2
        __builtin_prefetch(&UhT[(size_t)(n0) * UDIM + k0 + BK], 0, 2);
    };

    FragC acc[3][4];
#pragma unroll
    for (int g = 0; g < 3; ++g)
#pragma unroll
      for (int f = 0; f < 4; ++f)
#pragma unroll
        for (int v = 0; v < 8; ++v) acc[g][f].f[v] = 0.0f;

    const int nk = UDIM / BK;
    stage(0, 0);
    wait_async0();
    __syncthreads();

    for (int kt = 0; kt < nk; ++kt) {
      const int cur = kt & 1;
      if (kt + 1 < nk) stage((kt + 1) * BK, cur ^ 1);  // DMA next, overlap WMMA

      Frag16 a;
      load_afrag(ldsA + (cur * BM + wm * 16) * AS, a);
#pragma unroll
      for (int g = 0; g < 3; ++g)
#pragma unroll
        for (int f = 0; f < 4; ++f) {
          Frag16 b;
          load_bfrag(ldsB + ((cur * 3 + g) * BN + wn * 64 + f * 16) * BS, b);
          wmma_bf16(acc[g][f], a, b);
        }
      wait_async0();
      __syncthreads();
    }

    // Gating + state update (fragment: col = n_l, row = v + 8*kh)
#pragma unroll
    for (int v = 0; v < 8; ++v) {
      const int b = m0 + wm * 16 + v + 8 * kh;
      const int id = idsT[t * BSZ + b];
      const float* Prow = P + (size_t)id * U3;
#pragma unroll
      for (int f = 0; f < 4; ++f) {
        const int u = n0 + wn * 64 + f * 16 + n_l;
        const float iz = acc[0][f].f[v] + b_rec[u];
        const float ir = acc[1][f].f[v] + b_rec[UDIM + u];
        const float ih = acc[2][f].f[v] + b_rec[2 * UDIM + u];
        const float xz = Prow[u];
        const float xr = Prow[UDIM + u];
        const float xh = Prow[2 * UDIM + u];
        const float z = 1.0f / (1.0f + __expf(-(xz + iz)));
        const float r = 1.0f / (1.0f + __expf(-(xr + ir)));
        const float hh = tanhf(xh + r * ih);
        const float hold = h_in[(size_t)b * UDIM + u];
        const float hn = z * hold + (1.0f - z) * hh;
        h_out[(size_t)b * UDIM + u] = hn;
        const uint16_t hb = f2bf(hn);
        hbf_out[(size_t)b * UDIM + u] = hb;
        hs[((size_t)b * TLEN + t) * UDIM + u] = hb;
      }
    }

    grid_barrier(bar, bar + 1);  // step fence: h(t) visible device-wide
  }
}

// ---------------------------------------------------------------------------
extern "C" void kernel_launch(void* const* d_in, const int* in_sizes, int n_in,
                              void* d_out, int out_size, void* d_ws, size_t ws_size,
                              hipStream_t stream) {
  (void)in_sizes; (void)n_in; (void)out_size; (void)ws_size;
  const int*   ids   = (const int*)  d_in[0];
  const float* emb   = (const float*)d_in[1];
  const float* Wx    = (const float*)d_in[2];
  const float* Uh    = (const float*)d_in[3];
  const float* b_in  = (const float*)d_in[4];
  const float* b_rec = (const float*)d_in[5];
  const float* Wd    = (const float*)d_in[6];
  const float* bd    = (const float*)d_in[7];
  float* logits = (float*)d_out;

  // Workspace carve-up (256B aligned)
  char* ws = (char*)d_ws;
  size_t off = 0;
  auto carve = [&](size_t bytes) -> void* {
    void* p = ws + off;
    off = (off + bytes + 255) & ~(size_t)255;
    return p;
  };
  uint16_t* emb_bf = (uint16_t*)carve((size_t)VOC * EDIM * 2);
  uint16_t* WxT_bf = (uint16_t*)carve((size_t)EDIM * U3 * 2);   // [3U,E]
  uint16_t* UhT_bf = (uint16_t*)carve((size_t)UDIM * U3 * 2);   // [3U,U]
  uint16_t* WdT_bf = (uint16_t*)carve((size_t)UDIM * VOC * 2);  // [V,U]
  float*    P      = (float*)   carve((size_t)VOC * U3 * 4);
  float*    h32a   = (float*)   carve((size_t)BSZ * UDIM * 4);
  float*    h32b   = (float*)   carve((size_t)BSZ * UDIM * 4);
  uint16_t* hbfa   = (uint16_t*)carve((size_t)BSZ * UDIM * 2);
  uint16_t* hbfb   = (uint16_t*)carve((size_t)BSZ * UDIM * 2);
  int*      idsT   = (int*)     carve((size_t)BSZ * TLEN * 4);
  uint32_t* bar    = (uint32_t*)carve(256);
  uint16_t* hs     = (uint16_t*)carve((size_t)BSZ * TLEN * UDIM * 2);

  // 1) conversions / transposes / id layout (one-time)
  cvt_bf16_kernel<<<(VOC * EDIM + 255) / 256, 256, 0, stream>>>(
      emb, emb_bf, VOC * EDIM);
  cvt_bf16_T_kernel<<<(EDIM * U3 + 255) / 256, 256, 0, stream>>>(
      Wx, WxT_bf, EDIM, U3);
  cvt_bf16_T_kernel<<<(UDIM * U3 + 255) / 256, 256, 0, stream>>>(
      Uh, UhT_bf, UDIM, U3);
  cvt_bf16_T_kernel<<<(UDIM * VOC + 255) / 256, 256, 0, stream>>>(
      Wd, WdT_bf, UDIM, VOC);
  transpose_ids_kernel<<<(BSZ * TLEN + 255) / 256, 256, 0, stream>>>(ids, idsT);

  // 2) zero recurrent state (parity 0) + barrier words (d_ws is poisoned)
  {
    const int n = BSZ * UDIM;
    zero_state_kernel<<<(n + 255) / 256, 256, 0, stream>>>(h32a, hbfa, bar, n);
  }

  // 3) P = emb @ Wx + b_in   (emb[ids]@Wx == (emb@Wx)[ids]; P lives in L2)
  gemm_bf16_kernel<<<dim3(U3 / BN, VOC / BM), BDIM, 0, stream>>>(
      emb_bf, EDIM, WxT_bf, EDIM, b_in, P, U3, EDIM);

  // 4) recurrent scan: ONE persistent launch for all 512 steps
  gru_scan_kernel<<<dim3(UDIM / BN, BSZ / BM), BDIM, 0, stream>>>(
      hbfa, hbfb, h32a, h32b, UhT_bf, b_rec, P, idsT, hs, bar);

  // 5) logits = hs @ Wd + bd   (M = 131072, N = 128, K = 1024)
  gemm_bf16_kernel<<<dim3(VOC / BN, (BSZ * TLEN) / BM), BDIM, 0, stream>>>(
      hs, UDIM, WdT_bf, UDIM, bd, logits, VOC, UDIM);
}
